// GNNCoverageModel_66967130079866
// MI455X (gfx1250) — compile-verified, matrix operand order checked by reading, not run on hardware
//
#include <hip/hip_runtime.h>
#include <hip/hip_bf16.h>

// Problem constants (from reference setup_inputs)
#define NN 100000      // nodes
#define NE 1600000     // edges
#define INF 4          // input features
#define HC 128         // hidden channels
#define RR 20          // robots
#define OO 2           // output size (per robot)

typedef _Float16 v16h __attribute__((ext_vector_type(16)));
typedef _Float16 v8h  __attribute__((ext_vector_type(8)));
typedef float    v8f  __attribute__((ext_vector_type(8)));
typedef float    v2f  __attribute__((ext_vector_type(2)));

// ---------------------------------------------------------------------------
// 0) init: zero agg (N*H), deg=1.0 (self-loop), pooled=0, cnt=0
// ---------------------------------------------------------------------------
__global__ void k_init(float* __restrict__ agg, float* __restrict__ deg,
                       float* __restrict__ pooled, int* __restrict__ cnt) {
    long i = (long)blockIdx.x * blockDim.x + threadIdx.x;
    long total = (long)NN * HC;
    for (; i < total; i += (long)gridDim.x * blockDim.x) {
        agg[i] = 0.0f;
        if (i < NN)  deg[i] = 1.0f;       // self-loop contributes 1 to degree
        if (i < HC)  pooled[i] = 0.0f;
        if (i == 0)  *cnt = 0;
    }
}

// ---------------------------------------------------------------------------
// 1) degree accumulation over real edges
// ---------------------------------------------------------------------------
__global__ void k_deg(const int* __restrict__ dst, float* __restrict__ deg) {
    long i = (long)blockIdx.x * blockDim.x + threadIdx.x;
    if (i < NE) atomicAdd(&deg[dst[i]], 1.0f);
}

// 2) dinv = deg>0 ? rsqrt(deg) : 0   (in place)
__global__ void k_dinv(float* __restrict__ deg) {
    long i = (long)blockIdx.x * blockDim.x + threadIdx.x;
    if (i < NN) {
        float d = deg[i];
        deg[i] = (d > 0.0f) ? rsqrtf(d) : 0.0f;
    }
}

// ---------------------------------------------------------------------------
// 3) mm = x @ W1   ([N,4] @ [4,128])
//    Preferred: full-fp32 matrix pipe via V_WMMA_F32_16X16X4_F32 (K=4 native).
//    Fallback: plain fp32 VALU, one thread per output element.
// ---------------------------------------------------------------------------
#if __has_builtin(__builtin_amdgcn_wmma_f32_16x16x4_f32)
#define GEMM1_WMMA 1
__global__ void k_gemm1_wmma(const float* __restrict__ x, const float* __restrict__ W1,
                             float* __restrict__ mm) {
    int wave = (blockIdx.x * blockDim.x + threadIdx.x) >> 5;
    int lane = threadIdx.x & 31;
    int rowTile = wave >> 3;           // 8 column tiles (128/16)
    int colTile = wave & 7;
    if (rowTile >= NN / 16) return;
    int m = lane & 15;                 // A row / B column for this lane
    int g = lane >> 4;                 // lane group: K = g*2 + v
    // A fragment: x[(rowTile*16+m), g*2 + {0,1}]  (contiguous float2)
    v2f a = *(const v2f*)(x + (long)(rowTile * 16 + m) * INF + g * 2);
    // B fragment: W1[(g*2+v), colTile*16 + n]
    v2f b;
    b.x = W1[(g * 2 + 0) * HC + colTile * 16 + m];
    b.y = W1[(g * 2 + 1) * HC + colTile * 16 + m];
    v8f acc = {};
    acc = __builtin_amdgcn_wmma_f32_16x16x4_f32(
              false, a, false, b, (short)0, acc, false, false);
    int mBase = rowTile * 16 + g * 8;
    #pragma unroll
    for (int r = 0; r < 8; ++r)
        mm[(long)(mBase + r) * HC + colTile * 16 + m] = acc[r];
}
#else
#define GEMM1_WMMA 0
__global__ void k_gemm1(const float* __restrict__ x, const float* __restrict__ W1,
                        float* __restrict__ mm) {
    long i = (long)blockIdx.x * blockDim.x + threadIdx.x;
    long total = (long)NN * HC;
    if (i >= total) return;
    int n = (int)(i >> 7);
    int c = (int)(i & 127);
    const float* xr = x + (long)n * INF;
    float s = xr[0] * W1[0 * HC + c] + xr[1] * W1[1 * HC + c]
            + xr[2] * W1[2 * HC + c] + xr[3] * W1[3 * HC + c];
    mm[i] = s;
}
#endif

// ---------------------------------------------------------------------------
// 4) scatter: agg[dst] += h[src] * dinv[src]*dinv[dst]  over E edges + N loops
//    one wave per edge; lane handles 4 channels (float4 gather + 4 f32 atomics)
// ---------------------------------------------------------------------------
__global__ void k_scatter(const float* __restrict__ h,
                          const int* __restrict__ src, const int* __restrict__ dst,
                          const float* __restrict__ dinv, float* __restrict__ agg) {
    long t = (long)blockIdx.x * blockDim.x + threadIdx.x;
    long e = t >> 5;                 // edge id (wave per edge)
    int lane = (int)(t & 31);
    if (e >= (long)(NE + NN)) return;
    int s, d;
    if (e < NE) { s = src[e]; d = dst[e]; }
    else        { s = (int)(e - NE); d = s; }          // self-loop
    float norm = dinv[s] * dinv[d];
    long hb = (long)s * HC + lane * 4;
    long ab = (long)d * HC + lane * 4;
    __builtin_prefetch(&h[hb], 0, 0);
    float4 v = *(const float4*)&h[hb];
    atomicAdd(&agg[ab + 0], v.x * norm);
    atomicAdd(&agg[ab + 1], v.y * norm);
    atomicAdd(&agg[ab + 2], v.z * norm);
    atomicAdd(&agg[ab + 3], v.w * norm);
}

// ---------------------------------------------------------------------------
// 5) h1 = tanh(agg + b1) -> f16 ; re-zero agg for reuse by layer 2
// ---------------------------------------------------------------------------
__global__ void k_bias_tanh_f16(float* __restrict__ agg, const float* __restrict__ b,
                                _Float16* __restrict__ hf16) {
    long i = (long)blockIdx.x * blockDim.x + threadIdx.x;
    long total = (long)NN * HC;
    if (i >= total) return;
    int c = (int)(i & 127);
    float v = tanhf(agg[i] + b[c]);
    hf16[i] = (_Float16)v;
    agg[i] = 0.0f;                    // ready for the second scatter
}

// 6) W2t[n,k] = (f16) W2[k,n]   (transpose + downconvert, 128x128)
__global__ void k_w2t(const float* __restrict__ W2, _Float16* __restrict__ W2t) {
    int i = blockIdx.x * blockDim.x + threadIdx.x;
    if (i >= HC * HC) return;
    int k = i >> 7, n = i & 127;
    W2t[(long)n * HC + k] = (_Float16)W2[i];
}

// ---------------------------------------------------------------------------
// 7) mm = h1 @ W2 via v_wmma_f32_16x16x32_f16 (f32 accumulate, K=128 = 4 steps)
//    one wave per 16x16 output tile; A/B fragments are two 16B loads per step
// ---------------------------------------------------------------------------
__global__ void k_gemm2_wmma(const _Float16* __restrict__ A,   // [N,128] f16
                             const _Float16* __restrict__ Bt,  // [128,128] f16, W2^T
                             float* __restrict__ D) {          // [N,128] f32
    int wave = (blockIdx.x * blockDim.x + threadIdx.x) >> 5;
    int lane = threadIdx.x & 31;
    int rowTile = wave >> 3;          // 8 column tiles (128/16)
    int colTile = wave & 7;
    if (rowTile >= NN / 16) return;
    int m = lane & 15;                // A row / B column for this lane
    int g = lane >> 4;                // lane group (K-half select)
    const _Float16* arow = A  + (long)(rowTile * 16 + m) * HC;
    const _Float16* brow = Bt + (long)(colTile * 16 + m) * HC;
    v8f acc = {};
    #pragma unroll
    for (int kk = 0; kk < HC; kk += 32) {
        int kA = kk + g * 8;
        v8h alo = *(const v8h*)(arow + kA);
        v8h ahi = *(const v8h*)(arow + kA + 16);
        v8h blo = *(const v8h*)(brow + kA);
        v8h bhi = *(const v8h*)(brow + kA + 16);
        v16h a = __builtin_shufflevector(alo, ahi, 0,1,2,3,4,5,6,7,8,9,10,11,12,13,14,15);
        v16h b = __builtin_shufflevector(blo, bhi, 0,1,2,3,4,5,6,7,8,9,10,11,12,13,14,15);
        acc = __builtin_amdgcn_wmma_f32_16x16x32_f16(
                  false, a, false, b, (short)0, acc, false, false);
    }
    // D layout: lane n = lane&15 ; VGPR r -> M = r + g*8
    int nIdx = lane & 15;
    int mBase = rowTile * 16 + g * 8;
    #pragma unroll
    for (int r = 0; r < 8; ++r)
        D[(long)(mBase + r) * HC + colTile * 16 + nIdx] = acc[r];
}

// ---------------------------------------------------------------------------
// 8) pooled[c] += sum_n tanh(agg[n,c] + b2[c])  (per-block partial -> atomic)
// ---------------------------------------------------------------------------
#define NODES_PER_BLOCK 512
__global__ void k_tanh_pool(const float* __restrict__ agg, const float* __restrict__ b,
                            float* __restrict__ pooled) {
    int c = threadIdx.x;              // blockDim.x == 128
    int n0 = blockIdx.x * NODES_PER_BLOCK;
    int n1 = n0 + NODES_PER_BLOCK; if (n1 > NN) n1 = NN;
    float bias = b[c], local = 0.0f;
    for (int n = n0; n < n1; ++n)
        local += tanhf(agg[(long)n * HC + c] + bias);
    atomicAdd(&pooled[c], local);
}

// 9) in_size = #nodes with x[:,0]!=0 && x[:,1]!=0
__global__ void k_count(const float* __restrict__ x, int* __restrict__ cnt) {
    long i = (long)blockIdx.x * blockDim.x + threadIdx.x;
    if (i < NN) {
        if (x[i * INF + 0] != 0.0f && x[i * INF + 1] != 0.0f)
            atomicAdd(cnt, 1);
    }
}

// 10) out[j] = mask * (mean_pooled @ Wfc + bfc),  j in [0,40)
__global__ void k_final(const float* __restrict__ pooled, const float* __restrict__ Wfc,
                        const float* __restrict__ bfc, const int* __restrict__ cnt,
                        float* __restrict__ out) {
    int j = threadIdx.x;
    if (j >= RR * OO) return;
    float s = 0.0f;
    float inv = 1.0f / (float)NN;     // cnt(batch) == N, max(N,1)=N
    for (int c = 0; c < HC; ++c)
        s += (pooled[c] * inv) * Wfc[c * (RR * OO) + j];
    s += bfc[j];
    int r = j / OO;
    out[j] = (r < *cnt) ? s : 0.0f;
}

// ---------------------------------------------------------------------------
extern "C" void kernel_launch(void* const* d_in, const int* in_sizes, int n_in,
                              void* d_out, int out_size, void* d_ws, size_t ws_size,
                              hipStream_t stream) {
    const float* x    = (const float*)d_in[0];
    const int*   ei   = (const int*)d_in[1];     // [2,E]: src row, dst row
    // d_in[2] = batch (all zeros, single graph) -- unused
    const float* W1   = (const float*)d_in[3];
    const float* b1   = (const float*)d_in[4];
    const float* W2   = (const float*)d_in[5];
    const float* b2   = (const float*)d_in[6];
    const float* Wfc  = (const float*)d_in[7];
    const float* bfc  = (const float*)d_in[8];
    float* out = (float*)d_out;

    const int* src = ei;
    const int* dst = ei + NE;

    // workspace carve-up (all 256B aligned)
    char* ws = (char*)d_ws;
    size_t o = 0;
    float*    mm     = (float*)(ws + o);  o += (size_t)NN * HC * sizeof(float);   // 51.2MB
    float*    agg    = (float*)(ws + o);  o += (size_t)NN * HC * sizeof(float);   // 51.2MB
    _Float16* hf16   = (_Float16*)(ws + o); o += (size_t)NN * HC * sizeof(_Float16); // 25.6MB
    float*    dinv   = (float*)(ws + o);  o += (size_t)NN * sizeof(float);
    o = (o + 255) & ~(size_t)255;
    _Float16* W2t    = (_Float16*)(ws + o); o += (size_t)HC * HC * sizeof(_Float16);
    o = (o + 255) & ~(size_t)255;
    float*    pooled = (float*)(ws + o);  o += HC * sizeof(float);
    int*      cnt    = (int*)(ws + o);    o += 256;

    const long NH = (long)NN * HC;
    int blk = 256;
    int tiles = (NN / 16) * (HC / 16);            // 50000 waves, 8 waves/block

    // init accumulators (deterministic across graph replays)
    k_init<<<(int)((NH + blk - 1) / blk), blk, 0, stream>>>(agg, dinv, pooled, cnt);

    // normalization coefficients
    k_deg <<<(NE + blk - 1) / blk, blk, 0, stream>>>(dst, dinv);
    k_dinv<<<(NN + blk - 1) / blk, blk, 0, stream>>>(dinv);

    // ---- layer 1: x@W1 -> scatter -> tanh(+b1) -> f16
#if GEMM1_WMMA
    k_gemm1_wmma<<<tiles / 8, 256, 0, stream>>>(x, W1, mm);
#else
    k_gemm1<<<(int)((NH + blk - 1) / blk), blk, 0, stream>>>(x, W1, mm);
#endif
    long sThreads = (long)(NE + NN) * 32;
    k_scatter<<<(int)((sThreads + blk - 1) / blk), blk, 0, stream>>>(mm, src, dst, dinv, agg);
    k_bias_tanh_f16<<<(int)((NH + blk - 1) / blk), blk, 0, stream>>>(agg, b1, hf16);

    // ---- layer 2: h1@W2 (WMMA) -> scatter -> tanh(+b2) fused with mean pool
    k_w2t<<<(HC * HC + blk - 1) / blk, blk, 0, stream>>>(W2, W2t);
    k_gemm2_wmma<<<tiles / 8, 256, 0, stream>>>(hf16, W2t, mm);
    k_scatter<<<(int)((sThreads + blk - 1) / blk), blk, 0, stream>>>(mm, src, dst, dinv, agg);
    k_tanh_pool<<<(NN + NODES_PER_BLOCK - 1) / NODES_PER_BLOCK, HC, 0, stream>>>(agg, b2, pooled);

    // ---- head
    k_count<<<(NN + blk - 1) / blk, blk, 0, stream>>>(x, cnt);
    k_final<<<1, 64, 0, stream>>>(pooled, Wfc, bfc, cnt, out);
}
